// MemAttn_24068996727084
// MI455X (gfx1250) — compile-verified, matrix operand order checked by reading, not run on hardware
//
#include <hip/hip_runtime.h>
#include <hip/hip_bf16.h>
#include <math.h>

// MI455X (gfx1250): bf16 WMMA (16x16x32, fp32 accum) for all three GEMMs per
// hop. Loop-invariant operands (K, V^T, W) are pre-converted to bf16 once per
// call so hot loops issue pure bf16 b128 loads + v_wmma. B-fragment loads use
// a parity double-buffer (no copies) so tile t+1's loads issue before WMMA t.

typedef __attribute__((ext_vector_type(16))) __bf16 v16bf;
typedef __attribute__((ext_vector_type(8)))  __bf16 v8bf;
typedef __attribute__((ext_vector_type(4)))  __bf16 v4bf;
typedef __attribute__((ext_vector_type(8)))  float  v8f;

#define B_   16
#define LQ_  512
#define LK_  2048
#define D_   1024
#define SCALE_ 0.03125f   // 1/sqrt(1024)

// Fragment = two contiguous 8-element K-runs, 16 elements apart (CDNA5 ISA
// 7.12.2 layout for 16-bit A 16x32 / B 32x16, lane half selects K base).
__device__ __forceinline__ v16bf load_frag_f32(const float* __restrict__ p) {
  const float4* q = (const float4*)p;
  float4 f0 = q[0], f1 = q[1];
  float4 g0 = q[4], g1 = q[5];
  v16bf r;
  r[0]=(__bf16)f0.x; r[1]=(__bf16)f0.y; r[2]=(__bf16)f0.z; r[3]=(__bf16)f0.w;
  r[4]=(__bf16)f1.x; r[5]=(__bf16)f1.y; r[6]=(__bf16)f1.z; r[7]=(__bf16)f1.w;
  r[8]=(__bf16)g0.x; r[9]=(__bf16)g0.y; r[10]=(__bf16)g0.z; r[11]=(__bf16)g0.w;
  r[12]=(__bf16)g1.x; r[13]=(__bf16)g1.y; r[14]=(__bf16)g1.z; r[15]=(__bf16)g1.w;
  return r;
}

// Same fragment from bf16 memory (global or LDS): two 16B loads, zero cvt.
__device__ __forceinline__ v16bf load_frag_bf16(const __bf16* __restrict__ p) {
  v8bf lo = *(const v8bf*)p;
  v8bf hi = *(const v8bf*)(p + 16);
  v16bf r;
#pragma unroll
  for (int i = 0; i < 8; ++i) { r[i] = lo[i]; r[i + 8] = hi[i]; }
  return r;
}

__device__ __forceinline__ v8f wmma_bf16(v16bf a, v16bf b, v8f c) {
  return __builtin_amdgcn_wmma_f32_16x16x32_bf16(
      false, a, false, b, (short)0, c, false, false);
}

// ---------------------------------------------------------------------------
// Pre-pass A: elementwise f32 -> bf16 (K and W).
// ---------------------------------------------------------------------------
__global__ __launch_bounds__(256)
void k_cvt_bf16(const float* __restrict__ src, __bf16* __restrict__ dst, int n4) {
  for (int i = blockIdx.x * 256 + threadIdx.x; i < n4; i += gridDim.x * 256) {
    float4 v = ((const float4*)src)[i];
    v4bf o;
    o[0] = (__bf16)v.x; o[1] = (__bf16)v.y; o[2] = (__bf16)v.z; o[3] = (__bf16)v.w;
    ((v4bf*)dst)[i] = o;
  }
}

// ---------------------------------------------------------------------------
// Pre-pass B: V[b][k][d] f32 -> VT[b][d][k] bf16 (64x64 LDS tile transpose).
// ---------------------------------------------------------------------------
__global__ __launch_bounds__(256)
void k_transpose_v(const float* __restrict__ V, __bf16* __restrict__ VT) {
  __shared__ __bf16 Ls[64 * 65];
  const int kt  = blockIdx.x;    // 0..31  (LK/64)
  const int dt  = blockIdx.y;    // 0..15  (D/64)
  const int b   = blockIdx.z;
  const int tid = threadIdx.x;

  const float* Vb = V + (size_t)b * LK_ * D_;
#pragma unroll
  for (int i = 0; i < 4; ++i) {
    int j  = tid + i * 256;      // float4 id in 64x64 tile
    int kk = j >> 4;             // key within tile
    int d0 = (j & 15) * 4;       // d within tile
    float4 v = *(const float4*)(Vb + (size_t)(kt * 64 + kk) * D_ + dt * 64 + d0);
    Ls[(d0 + 0) * 65 + kk] = (__bf16)v.x;
    Ls[(d0 + 1) * 65 + kk] = (__bf16)v.y;
    Ls[(d0 + 2) * 65 + kk] = (__bf16)v.z;
    Ls[(d0 + 3) * 65 + kk] = (__bf16)v.w;
  }
  __syncthreads();
  __bf16* Ob = VT + (size_t)b * D_ * LK_;
#pragma unroll
  for (int i = 0; i < 4; ++i) {
    int j  = tid + i * 256;
    int dd = j >> 4;
    int k0 = (j & 15) * 4;
    v4bf o;
    o[0] = Ls[dd * 65 + k0 + 0]; o[1] = Ls[dd * 65 + k0 + 1];
    o[2] = Ls[dd * 65 + k0 + 2]; o[3] = Ls[dd * 65 + k0 + 3];
    *(v4bf*)(Ob + (size_t)(dt * 64 + dd) * LK_ + kt * 64 + k0) = o;
  }
}

// ---------------------------------------------------------------------------
// Kernel 1: S = scale * Qhop K^T for a 16-query tile, full-row softmax,
// write normalized attn (f32). WG = 8 waves; wave w owns 256 keys (16 tiles).
// ---------------------------------------------------------------------------
__global__ __launch_bounds__(256)
void k_scores_softmax(const float* __restrict__ Qhop,
                      const __bf16* __restrict__ Kb16,
                      float* __restrict__ attn) {
  __shared__ __bf16 Qs[16][D_];   // 32KB query tile as bf16
  __shared__ float  red[8][16];

  const int qt   = blockIdx.x;
  const int b    = blockIdx.y;
  const int tid  = threadIdx.x;
  const int wave = tid >> 5;
  const int lane = tid & 31;
  const int half = lane >> 4;
  const int l16  = lane & 15;

  const float* Qblk = Qhop + ((size_t)b * LQ_ + (size_t)qt * 16) * D_;
  for (int i = tid; i < (16 * D_) / 4; i += 256) {
    float4 v = ((const float4*)Qblk)[i];
    int r = (i * 4) / D_, c = (i * 4) % D_;
    Qs[r][c + 0] = (__bf16)v.x; Qs[r][c + 1] = (__bf16)v.y;
    Qs[r][c + 2] = (__bf16)v.z; Qs[r][c + 3] = (__bf16)v.w;
  }
  __syncthreads();

  v8f acc[16];
#pragma unroll
  for (int t = 0; t < 16; ++t)
#pragma unroll
    for (int i = 0; i < 8; ++i) acc[t][i] = 0.f;

  const int col0 = wave * 256;
  const __bf16* Kb = Kb16 + (size_t)b * LK_ * D_;

  for (int s = 0; s < D_ / 32; ++s) {
    const int kb = s * 32;
    // Prefetch next K-step for this wave's key range (global_prefetch_b8).
    __builtin_prefetch((const void*)(Kb + (size_t)(col0 + lane * 8) * D_ + kb + 32), 0, 1);
    v16bf a = load_frag_bf16(&Qs[l16][kb + half * 8]);
    const __bf16* kp = Kb + (size_t)(col0 + l16) * D_ + kb + half * 8;
    // Parity double-buffer (no copies): tile t+1 loads issue before WMMA t.
    v16bf buf[2];
    buf[0] = load_frag_bf16(kp);
#pragma unroll
    for (int t = 0; t < 16; ++t) {
      if (t + 1 < 16)
        buf[(t + 1) & 1] = load_frag_bf16(kp + (size_t)(t + 1) * 16 * D_);
      acc[t] = wmma_bf16(a, buf[t & 1], acc[t]);
    }
  }

#pragma unroll
  for (int t = 0; t < 16; ++t)
#pragma unroll
    for (int v = 0; v < 8; ++v) acc[t][v] *= SCALE_;

  float rmax[8];
#pragma unroll
  for (int v = 0; v < 8; ++v) {
    float m = acc[0][v];
#pragma unroll
    for (int t = 1; t < 16; ++t) m = fmaxf(m, acc[t][v]);
#pragma unroll
    for (int off = 1; off < 16; off <<= 1) m = fmaxf(m, __shfl_xor(m, off, 32));
    rmax[v] = m;
  }
  if (l16 == 0) {
#pragma unroll
    for (int v = 0; v < 8; ++v) red[wave][half * 8 + v] = rmax[v];
  }
  __syncthreads();
  float gmax[8];
#pragma unroll
  for (int v = 0; v < 8; ++v) {
    float m = red[0][half * 8 + v];
#pragma unroll
    for (int w = 1; w < 8; ++w) m = fmaxf(m, red[w][half * 8 + v]);
    gmax[v] = m;
  }
  __syncthreads();

  float rsum[8];
#pragma unroll
  for (int v = 0; v < 8; ++v) rsum[v] = 0.f;
#pragma unroll
  for (int t = 0; t < 16; ++t)
#pragma unroll
    for (int v = 0; v < 8; ++v) {
      float p = __expf(acc[t][v] - gmax[v]);
      acc[t][v] = p;
      rsum[v] += p;
    }
#pragma unroll
  for (int v = 0; v < 8; ++v) {
    float s = rsum[v];
#pragma unroll
    for (int off = 1; off < 16; off <<= 1) s += __shfl_xor(s, off, 32);
    rsum[v] = s;
  }
  if (l16 == 0) {
#pragma unroll
    for (int v = 0; v < 8; ++v) red[wave][half * 8 + v] = rsum[v];
  }
  __syncthreads();
  float ginv[8];
#pragma unroll
  for (int v = 0; v < 8; ++v) {
    float s = 0.f;
#pragma unroll
    for (int w = 0; w < 8; ++w) s += red[w][half * 8 + v];
    ginv[v] = 1.f / s;
  }

  float* arow = attn + ((size_t)b * LQ_ + (size_t)qt * 16) * LK_;
#pragma unroll
  for (int t = 0; t < 16; ++t) {
    const int col = col0 + t * 16 + l16;
#pragma unroll
    for (int v = 0; v < 8; ++v)
      arow[(size_t)(half * 8 + v) * LK_ + col] = acc[t][v] * ginv[v];
  }
}

// ---------------------------------------------------------------------------
// Kernel 2: R = attn @ V using V^T bf16 (fragment-ready). No LDS, no barriers.
// WG = 4 waves computes 16x512; each wave 8 tiles. R stored bf16.
// ---------------------------------------------------------------------------
__global__ __launch_bounds__(128)
void k_attn_v(const float* __restrict__ attn,
              const __bf16* __restrict__ VT,
              __bf16* __restrict__ Rb16) {
  const int nt   = blockIdx.x;   // 0..1 (512-wide N tile over D)
  const int qt   = blockIdx.y;   // 0..31
  const int b    = blockIdx.z;
  const int tid  = threadIdx.x;
  const int wave = tid >> 5;
  const int lane = tid & 31;
  const int half = lane >> 4;
  const int l16  = lane & 15;
  const int nbase = nt * 512;

  v8f acc[8];
#pragma unroll
  for (int t = 0; t < 8; ++t)
#pragma unroll
    for (int i = 0; i < 8; ++i) acc[t][i] = 0.f;

  const float* Ar = attn + ((size_t)b * LQ_ + (size_t)qt * 16 + l16) * LK_;
  const __bf16* Vt = VT + (size_t)b * D_ * LK_;

  for (int s = 0; s < LK_ / 32; ++s) {
    const int kb = s * 32;
    v16bf a = load_frag_f32(Ar + kb + half * 8);
    const __bf16* vp = Vt + (size_t)(nbase + wave * 128 + l16) * LK_ + kb + half * 8;
    v16bf buf[2];
    buf[0] = load_frag_bf16(vp);
#pragma unroll
    for (int t = 0; t < 8; ++t) {
      if (t + 1 < 8)
        buf[(t + 1) & 1] = load_frag_bf16(vp + (size_t)(t + 1) * 16 * LK_);
      acc[t] = wmma_bf16(a, buf[t & 1], acc[t]);
    }
  }

  __bf16* Rr = Rb16 + ((size_t)b * LQ_ + (size_t)qt * 16) * D_;
#pragma unroll
  for (int t = 0; t < 8; ++t) {
    const int n = nbase + wave * 128 + t * 16 + l16;
#pragma unroll
    for (int v = 0; v < 8; ++v)
      Rr[(size_t)(half * 8 + v) * D_ + n] = (__bf16)acc[t][v];
  }
}

// ---------------------------------------------------------------------------
// Kernel 3: out = tanh([Qhop | R] @ W^T + b). A: f32 Qhop (cvt) then bf16 R
// (direct). B: pre-converted W rows (fragment-ready). Wave owns 8 tiles.
// ---------------------------------------------------------------------------
__global__ __launch_bounds__(128)
void k_linear_tanh(const float* __restrict__ Qhop,
                   const __bf16* __restrict__ Rb16,
                   const __bf16* __restrict__ Wb16,
                   const float* __restrict__ bias,
                   float* __restrict__ out) {
  const int nt   = blockIdx.x;   // 0..1
  const int mt   = blockIdx.y;   // 0..511
  const int tid  = threadIdx.x;
  const int wave = tid >> 5;
  const int lane = tid & 31;
  const int half = lane >> 4;
  const int l16  = lane & 15;
  const int nbase = nt * 512;

  v8f acc[8];
#pragma unroll
  for (int t = 0; t < 8; ++t)
#pragma unroll
    for (int i = 0; i < 8; ++i) acc[t][i] = 0.f;

  const size_t row = (size_t)mt * 16 + l16;
  const float*  Arq = Qhop + row * D_;
  const __bf16* Arr = Rb16 + row * D_;

  for (int s = 0; s < 64; ++s) {
    const int kb = s * 32;
    v16bf a = (s < 32) ? load_frag_f32(Arq + kb + half * 8)
                       : load_frag_bf16(Arr + (kb - D_) + half * 8);
    const __bf16* wp = Wb16 + (size_t)(nbase + wave * 128 + l16) * (2 * D_) + kb + half * 8;
    v16bf buf[2];
    buf[0] = load_frag_bf16(wp);
#pragma unroll
    for (int t = 0; t < 8; ++t) {
      if (t + 1 < 8)
        buf[(t + 1) & 1] = load_frag_bf16(wp + (size_t)(t + 1) * 16 * (2 * D_));
      acc[t] = wmma_bf16(a, buf[t & 1], acc[t]);
    }
  }

  float* orow = out + (size_t)mt * 16 * D_;
#pragma unroll
  for (int t = 0; t < 8; ++t) {
    const int n = nbase + wave * 128 + t * 16 + l16;
    const float bv = bias[n];
#pragma unroll
    for (int v = 0; v < 8; ++v)
      orow[(size_t)(half * 8 + v) * D_ + n] = tanhf(acc[t][v] + bv);
  }
}

// ---------------------------------------------------------------------------
extern "C" void kernel_launch(void* const* d_in, const int* in_sizes, int n_in,
                              void* d_out, int out_size, void* d_ws, size_t ws_size,
                              hipStream_t stream) {
  (void)in_sizes; (void)n_in; (void)out_size; (void)ws_size;
  const float* Q    = (const float*)d_in[0];
  const float* K    = (const float*)d_in[1];
  const float* V    = (const float*)d_in[2];
  const float* W    = (const float*)d_in[3];
  const float* bias = (const float*)d_in[4];

  const size_t QN  = (size_t)B_ * LQ_ * D_;   // 8,388,608
  const size_t KVN = (size_t)B_ * LK_ * D_;   // 33,554,432
  const size_t WN  = (size_t)D_ * 2 * D_;     // 2,097,152

  float* outQ = (float*)d_out;
  float* attn = outQ + QN;

  __bf16* Kb16 = (__bf16*)d_ws;
  __bf16* VT   = Kb16 + KVN;
  __bf16* Wb16 = VT + KVN;
  __bf16* Rb16 = Wb16 + WN;
  float*  bufA = (float*)(Rb16 + QN);

  // Pre-convert loop-invariant operands (identical rounding to cvt-at-load).
  k_cvt_bf16<<<4096, 256, 0, stream>>>(K, Kb16, (int)(KVN / 4));
  k_cvt_bf16<<<1024, 256, 0, stream>>>(W, Wb16, (int)(WN / 4));
  k_transpose_v<<<dim3(LK_ / 64, D_ / 64, B_), 256, 0, stream>>>(V, VT);

  // Q-buffer chain: hop0 Q->outQ, hop1 outQ->bufA, hop2 bufA->outQ.
  const float* qcur = Q;
  for (int hop = 0; hop < 3; ++hop) {
    k_scores_softmax<<<dim3(LQ_ / 16, B_), 256, 0, stream>>>(qcur, Kb16, attn);
    k_attn_v<<<dim3(2, LQ_ / 16, B_), 128, 0, stream>>>(attn, VT, Rb16);
    float* qnext = (hop == 1) ? bufA : outQ;
    k_linear_tanh<<<dim3(2, (B_ * LQ_) / 16), 128, 0, stream>>>(
        qcur, Rb16, Wb16, bias, qnext);
    qcur = qnext;
  }
}
